// GAT_32701880991773
// MI455X (gfx1250) — matrix operand order repaired and checked by reading, not compiled, not run on hardware
//
#include <hip/hip_runtime.h>
#include <hip/hip_bf16.h>
#include <math.h>

typedef _Float16 f16;
typedef __attribute__((ext_vector_type(16))) _Float16 v16h;
typedef __attribute__((ext_vector_type(8)))  _Float16 v8h;
typedef __attribute__((ext_vector_type(8)))  float    v8f;

#define HEADS    8
#define HIDDEN   16
#define NN       20000
#define NE       160000
#define NG       32
#define NFF      512
#define NL       5

// ------------------------- generic fills -------------------------
__global__ void k_fill_f32(float* p, float v, long n) {
    long i = (long)blockIdx.x * blockDim.x + threadIdx.x;
    if (i < n) p[i] = v;
}
__global__ void k_fill_u32(unsigned* p, unsigned v, long n) {
    long i = (long)blockIdx.x * blockDim.x + threadIdx.x;
    if (i < n) p[i] = v;
}

// ------------------------- embeddings -------------------------
__global__ void k_emb_vert(const float* __restrict__ x, const float* __restrict__ W,
                           float* __restrict__ h) {
    long i = (long)blockIdx.x * blockDim.x + threadIdx.x;
    if (i >= (long)NN * HIDDEN) return;
    int c = (int)(i % HIDDEN); long n = i / HIDDEN;
    float acc = 0.f;
    for (int j = 0; j < 13; ++j) acc += x[n * 13 + j] * W[j * HIDDEN + c];
    h[i] = acc;
}
__global__ void k_emb_edge(const float* __restrict__ ea, const float* __restrict__ W,
                           float* __restrict__ e) {
    long i = (long)blockIdx.x * blockDim.x + threadIdx.x;
    if (i >= (long)NE * HIDDEN) return;
    int c = (int)(i % HIDDEN); long n = i / HIDDEN;
    float acc = 0.f;
    for (int j = 0; j < 4; ++j) acc += ea[n * 4 + j] * W[j * HIDDEN + c];
    e[i] = acc;
}

// ------------------------- f16 packing -------------------------
// [M, srcC] f32 -> [M, dstC] f16, zero-padded columns
__global__ void k_f32_to_f16_pad(const float* __restrict__ src, int srcC,
                                 f16* __restrict__ dst, int dstC, long total) {
    long i = (long)blockIdx.x * blockDim.x + threadIdx.x;
    if (i >= total) return;
    int c = (int)(i % dstC); long r = i / dstC;
    dst[i] = (c < srcC) ? (f16)src[r * srcC + c] : (f16)0.f;
}
// [N, H*dsrc] f32 -> [N, H*dpad] f16 (per-head zero padding)
__global__ void k_pack_heads_f16(const float* __restrict__ src, f16* __restrict__ dst,
                                 int dsrc, int dpad, long total) {
    long i = (long)blockIdx.x * blockDim.x + threadIdx.x;
    if (i >= total) return;
    int c = (int)(i % dpad); long t = i / dpad;
    int h = (int)(t % HEADS); long n = t / HEADS;
    dst[i] = (c < dsrc) ? (f16)src[(n * HEADS + h) * (long)dsrc + c] : (f16)0.f;
}
// W [K, N] f32 (row stride ldw) -> Bt [N, Kpad] f16 (transposed, K zero-padded)
__global__ void k_repack_wT(const float* __restrict__ W, int ldw, int K,
                            f16* __restrict__ dst, int Kpad, long total) {
    long i = (long)blockIdx.x * blockDim.x + threadIdx.x;
    if (i >= total) return;
    int k = (int)(i % Kpad); long n = i / Kpad;
    dst[i] = (k < K) ? (f16)W[(long)k * ldw + n] : (f16)0.f;
}
// We [16, H*dout] -> per-head row slices: dst[(h*16+j)*doutPad + c] = We[j, h*dout+c]
__global__ void k_repack_WeP(const float* __restrict__ We, int dout, int doutPad,
                             f16* __restrict__ dst) {
    int total = HEADS * 16 * doutPad;
    int i = blockIdx.x * blockDim.x + threadIdx.x;
    if (i >= total) return;
    int c = i % doutPad; int t = i / doutPad; int j = t % 16; int h = t / 16;
    int N8 = HEADS * dout;
    dst[i] = (c < dout) ? (f16)We[j * N8 + h * dout + c] : (f16)0.f;
}
// We [16, H*dout] -> per-head transposed: dst[(h*dout+c)*32 + j] = We[j, h*dout+c]
__global__ void k_repack_WeT(const float* __restrict__ We, int dout, f16* __restrict__ dst) {
    int total = HEADS * dout * 32;
    int i = blockIdx.x * blockDim.x + threadIdx.x;
    if (i >= total) return;
    int j = i % 32; int t = i / 32; int c = t % dout; int h = t / dout;
    int N8 = HEADS * dout;
    dst[i] = (j < 16) ? (f16)We[j * N8 + h * dout + c] : (f16)0.f;
}

// ------------------------- WMMA GEMM -------------------------
// C[16mt..+16, 16nt..+16] = A[M,K]f16 * Bt[N,K]f16^T (+ bias[col]) (+ Cin tile)
// K must be a multiple of 32. One wave per tile.
__global__ __launch_bounds__(32)
void k_wmma_gemm(const f16* __restrict__ A, int lda,
                 const f16* __restrict__ Bt, int ldb,
                 const float* __restrict__ bias,
                 const float* __restrict__ Cin, int ldcin,
                 float* __restrict__ C, int ldc, int K) {
    const int lane = threadIdx.x;
    const int mt = blockIdx.x, nt = blockIdx.y;
    const int hi = lane >> 4;       // 0 or 1
    const int l16 = lane & 15;

    v8f acc = {};
    const int col = nt * 16 + l16;
    const int rowbase = mt * 16 + hi * 8;
    if (Cin) {
#pragma unroll
        for (int r = 0; r < 8; ++r) acc[r] = Cin[(long)(rowbase + r) * ldcin + col];
    }
    // A fragment: lane<16 holds row M=l16, K 0..15 contiguous; lane>=16: K 16..31
    const f16* arow = A + (long)(mt * 16 + l16) * lda + (hi ? 16 : 0);
    // B fragment: lane<16 holds col N=l16, K 0..15 contiguous; lane>=16: K 16..31
    const f16* brow = Bt + (long)(nt * 16 + l16) * ldb + (hi ? 16 : 0);

    for (int k0 = 0; k0 < K; k0 += 32) {
        v8h alo = *(const v8h*)(arow + k0);
        v8h ahi = *(const v8h*)(arow + k0 + 8);
        v16h a;
#pragma unroll
        for (int t = 0; t < 8; ++t) { a[t] = alo[t]; a[t + 8] = ahi[t]; }
        v16h b = *(const v16h*)(brow + k0);
        acc = __builtin_amdgcn_wmma_f32_16x16x32_f16(false, a, false, b,
                                                     (short)0, acc, false, false);
    }
    float bv = bias ? bias[col] : 0.f;
#pragma unroll
    for (int r = 0; r < 8; ++r)
        C[(long)(rowbase + r) * ldc + col] = acc[r] + bv;
}

// ------------------------- attention (scalar side) -------------------------
__device__ __forceinline__ unsigned fkey(float f) {
    unsigned u = __float_as_uint(f);
    return (u & 0x80000000u) ? ~u : (u | 0x80000000u);
}
__device__ __forceinline__ float fdecode(unsigned k) {
    unsigned u = (k & 0x80000000u) ? (k & 0x7FFFFFFFu) : ~k;
    return __uint_as_float(u);
}

__global__ void k_alpha(const float* __restrict__ q, const float* __restrict__ kk,
                        const float* __restrict__ Qe, const float* __restrict__ eemb,
                        const int* __restrict__ src, const int* __restrict__ dst,
                        float* __restrict__ alpha, unsigned* __restrict__ mkey,
                        int dout, float scale) {
    long i = (long)blockIdx.x * blockDim.x + threadIdx.x;
    if (i >= (long)NE * HEADS) return;
    int h = (int)(i % HEADS); long e = i / HEADS;
    int s = src[e], d = dst[e];
    int N8 = HEADS * dout;
    const float* qp = q + (long)d * N8 + h * dout;
    const float* kp = kk + (long)s * N8 + h * dout;
    float acc = 0.f;
    for (int c = 0; c < dout; ++c) acc += qp[c] * kp[c];
    const float* ep = eemb + e * (long)HIDDEN;
    const float* Qp = Qe + (long)d * (HEADS * HIDDEN) + h * HIDDEN;
    for (int j = 0; j < HIDDEN; ++j) acc += ep[j] * Qp[j];
    float a = acc * scale;
    alpha[i] = a;
    atomicMax(&mkey[d * HEADS + h], fkey(a));
}

__global__ void k_expsum(float* __restrict__ alpha, const unsigned* __restrict__ mkey,
                         float* __restrict__ denom, const int* __restrict__ dst) {
    long i = (long)blockIdx.x * blockDim.x + threadIdx.x;
    if (i >= (long)NE * HEADS) return;
    int h = (int)(i % HEADS); long e = i / HEADS;
    int d = dst[e];
    float m = fdecode(mkey[d * HEADS + h]);
    float a = __expf(alpha[i] - m);
    alpha[i] = a;
    atomicAdd(&denom[d * HEADS + h], a);
}

__global__ void k_scatter(const float* __restrict__ alpha, const float* __restrict__ denom,
                          const float* __restrict__ v, const float* __restrict__ eemb,
                          const int* __restrict__ src, const int* __restrict__ dst,
                          float* __restrict__ out_v, float* __restrict__ S, int dout) {
    long i = (long)blockIdx.x * blockDim.x + threadIdx.x;
    if (i >= (long)NE * HEADS) return;
    int h = (int)(i % HEADS); long e = i / HEADS;
    int s = src[e], d = dst[e];
    float a = alpha[i] / (denom[d * HEADS + h] + 1e-16f);
    int N8 = HEADS * dout;
    const float* vp = v + (long)s * N8 + h * dout;
    float* op = out_v + (long)d * N8 + h * dout;
    for (int c = 0; c < dout; ++c) atomicAdd(&op[c], a * vp[c]);
    const float* ep = eemb + e * (long)HIDDEN;
    float* Sp = S + (long)d * (HEADS * HIDDEN) + h * HIDDEN;
    for (int j = 0; j < HIDDEN; ++j) atomicAdd(&Sp[j], a * ep[j]);
}

// h_out = relu(mean_h(out_v) + skip) + res
__global__ void k_finalize(const float* __restrict__ out_v, const float* __restrict__ skip,
                           const float* __restrict__ res, float* __restrict__ hout, int dout) {
    long i = (long)blockIdx.x * blockDim.x + threadIdx.x;
    if (i >= (long)NN * dout) return;
    int c = (int)(i % dout); long n = i / dout;
    int N8 = HEADS * dout;
    float acc = 0.f;
    for (int h = 0; h < HEADS; ++h) acc += out_v[n * N8 + h * dout + c];
    acc = acc * (1.0f / HEADS) + skip[i];
    hout[i] = fmaxf(acc, 0.f) + res[i];
}

// ------------------------- pooling + head -------------------------
__global__ void k_pool(const float* __restrict__ h, const int* __restrict__ batch,
                       float* __restrict__ g, float* __restrict__ cnt, int dF) {
    long i = (long)blockIdx.x * blockDim.x + threadIdx.x;
    if (i >= (long)NN * dF) return;
    int c = (int)(i % dF); long n = i / dF;
    int b = batch[n];
    atomicAdd(&g[b * dF + c], h[i]);
    if (c == 0) atomicAdd(&cnt[b], 1.0f);
}
__global__ void k_head1(const float* __restrict__ g, const float* __restrict__ cnt,
                        const float* __restrict__ W1, const float* __restrict__ b1,
                        float* __restrict__ ff, int dF) {
    int i = blockIdx.x * blockDim.x + threadIdx.x;
    if (i >= NG * NFF) return;
    int f = i % NFF; int gi = i / NFF;
    float inv = 1.0f / fmaxf(cnt[gi], 1.0f);
    float acc = b1[f];
    for (int c = 0; c < dF; ++c) acc += (g[gi * dF + c] * inv) * W1[c * NFF + f];
    ff[i] = 0.5f * acc * (1.0f + erff(acc * 0.70710678118654752f));  // exact GELU
}
__global__ void k_head2(const float* __restrict__ ff, const float* __restrict__ W2,
                        const float* __restrict__ b2, float* __restrict__ out) {
    int gi = blockIdx.x * blockDim.x + threadIdx.x;
    if (gi >= NG) return;
    float acc = b2[0];
    for (int f = 0; f < NFF; ++f) acc += ff[gi * NFF + f] * W2[f];
    out[gi] = acc;
}

// ------------------------- host driver -------------------------
#define L1D(kern, n, ...) \
    kern<<<dim3((unsigned)(((long)(n) + 255) / 256)), dim3(256), 0, stream>>>(__VA_ARGS__)

extern "C" void kernel_launch(void* const* d_in, const int* in_sizes, int n_in,
                              void* d_out, int out_size, void* d_ws, size_t ws_size,
                              hipStream_t stream) {
    (void)in_sizes; (void)n_in; (void)out_size; (void)ws_size;
    const float* x         = (const float*)d_in[0];
    const int*   eidx      = (const int*)d_in[1];
    const float* edge_attr = (const float*)d_in[2];
    const int*   batch     = (const int*)d_in[3];
    const float* vertW     = (const float*)d_in[4];
    const float* edgeW     = (const float*)d_in[5];
    const float* headW1    = (const float*)d_in[56];
    const float* headb1    = (const float*)d_in[57];
    const float* headW2    = (const float*)d_in[58];
    const float* headb2    = (const float*)d_in[59];
    float* out = (float*)d_out;

    static const int widths[6] = {1, 1, 2, 3, 5, 7};

    // bump allocator over workspace (256B aligned)
    char* ws = (char*)d_ws;
    size_t off = 0;
    auto alloc = [&](size_t bytes) -> char* {
        off = (off + 255) & ~(size_t)255;
        char* p = ws + off; off += bytes; return p;
    };
    float*    hA    = (float*)alloc((size_t)NN * 112 * 4);
    float*    hB    = (float*)alloc((size_t)NN * 112 * 4);
    float*    eemb  = (float*)alloc((size_t)NE * 16 * 4);
    float*    qb    = (float*)alloc((size_t)NN * 896 * 4);
    float*    kb    = (float*)alloc((size_t)NN * 896 * 4);
    float*    vb    = (float*)alloc((size_t)NN * 896 * 4);
    float*    out_v = (float*)alloc((size_t)NN * 896 * 4);
    float*    Qe    = (float*)alloc((size_t)NN * 128 * 4);
    float*    S     = (float*)alloc((size_t)NN * 128 * 4);
    float*    skip  = (float*)alloc((size_t)NN * 112 * 4);
    float*    res   = (float*)alloc((size_t)NN * 112 * 4);
    float*    alpha = (float*)alloc((size_t)NE * 8 * 4);
    float*    denom = (float*)alloc((size_t)NN * 8 * 4);
    unsigned* mkey  = (unsigned*)alloc((size_t)NN * 8 * 4);
    float*    g     = (float*)alloc((size_t)NG * 112 * 4);
    float*    cnt   = (float*)alloc((size_t)NG * 4);
    float*    ffb   = (float*)alloc((size_t)NG * NFF * 4);
    f16*      hf16  = (f16*)alloc((size_t)NN * 96 * 2);
    f16*      qf16  = (f16*)alloc((size_t)NN * 1024 * 2);
    f16*      Sf16  = (f16*)alloc((size_t)NN * 256 * 2);
    f16*      WqT   = (f16*)alloc((size_t)896 * 96 * 2);
    f16*      WkT   = (f16*)alloc((size_t)896 * 96 * 2);
    f16*      WvT   = (f16*)alloc((size_t)896 * 96 * 2);
    f16*      WskT  = (f16*)alloc((size_t)112 * 96 * 2);
    f16*      WrT   = (f16*)alloc((size_t)112 * 96 * 2);
    f16*      WeP   = (f16*)alloc((size_t)8 * 16 * 128 * 2);
    f16*      WeT   = (f16*)alloc((size_t)8 * 112 * 32 * 2);

    const int* src  = eidx;
    const int* dstp = eidx + NE;

    // input embeddings
    L1D(k_emb_vert, (long)NN * HIDDEN, x, vertW, hA);
    L1D(k_emb_edge, (long)NE * HIDDEN, edge_attr, edgeW, eemb);

    for (int i = 0; i < NL; ++i) {
        const int din = widths[i] * 16, dout = widths[i + 1] * 16;
        const int dinPad = (din + 31) & ~31, doutPad = (dout + 31) & ~31;
        const int N8 = HEADS * dout;
        const int base = 6 + 10 * i;
        const float* Wq    = (const float*)d_in[base + 0];
        const float* bq    = (const float*)d_in[base + 1];
        const float* Wk    = (const float*)d_in[base + 2];
        const float* bk    = (const float*)d_in[base + 3];
        const float* Wv    = (const float*)d_in[base + 4];
        const float* bv    = (const float*)d_in[base + 5];
        const float* We    = (const float*)d_in[base + 6];
        const float* Wskip = (const float*)d_in[base + 7];
        const float* bskip = (const float*)d_in[base + 8];
        const float* Wres  = (const float*)d_in[base + 9];

        // weight repacks (f32 -> padded f16, B-transposed layouts)
        L1D(k_repack_wT, (long)N8 * dinPad, Wq, N8, din, WqT, dinPad, (long)N8 * dinPad);
        L1D(k_repack_wT, (long)N8 * dinPad, Wk, N8, din, WkT, dinPad, (long)N8 * dinPad);
        L1D(k_repack_wT, (long)N8 * dinPad, Wv, N8, din, WvT, dinPad, (long)N8 * dinPad);
        L1D(k_repack_wT, (long)dout * dinPad, Wskip, dout, din, WskT, dinPad, (long)dout * dinPad);
        L1D(k_repack_wT, (long)dout * dinPad, Wres, dout, din, WrT, dinPad, (long)dout * dinPad);
        L1D(k_repack_WeP, HEADS * 16 * doutPad, We, dout, doutPad, WeP);
        L1D(k_repack_WeT, HEADS * dout * 32, We, dout, WeT);

        // node features -> f16 (K padded)
        L1D(k_f32_to_f16_pad, (long)NN * dinPad, hA, din, hf16, dinPad, (long)NN * dinPad);

        // dense projections via WMMA
        dim3 gq(NN / 16, N8 / 16), gs(NN / 16, dout / 16);
        k_wmma_gemm<<<gq, 32, 0, stream>>>(hf16, dinPad, WqT, dinPad, bq, nullptr, 0, qb, N8, dinPad);
        k_wmma_gemm<<<gq, 32, 0, stream>>>(hf16, dinPad, WkT, dinPad, bk, nullptr, 0, kb, N8, dinPad);
        k_wmma_gemm<<<gq, 32, 0, stream>>>(hf16, dinPad, WvT, dinPad, bv, nullptr, 0, vb, N8, dinPad);
        k_wmma_gemm<<<gs, 32, 0, stream>>>(hf16, dinPad, WskT, dinPad, bskip, nullptr, 0, skip, dout, dinPad);
        k_wmma_gemm<<<gs, 32, 0, stream>>>(hf16, dinPad, WrT, dinPad, nullptr, nullptr, 0, res, dout, dinPad);

        // Qe[n,h,:] = We_h^T q[n,h,:]  (per-head WMMA GEMM, N=16, K=doutPad)
        L1D(k_pack_heads_f16, (long)NN * HEADS * doutPad, qb, qf16, dout, doutPad,
            (long)NN * HEADS * doutPad);
        for (int h = 0; h < HEADS; ++h)
            k_wmma_gemm<<<dim3(NN / 16, 1), 32, 0, stream>>>(
                qf16 + h * doutPad, HEADS * doutPad, WeP + h * 16 * doutPad, doutPad,
                nullptr, nullptr, 0, Qe + h * 16, HEADS * HIDDEN, doutPad);

        // attention: segment softmax + scatter
        L1D(k_fill_u32, (long)NN * 8, mkey, 0u, (long)NN * 8);
        L1D(k_fill_f32, (long)NN * 8, denom, 0.f, (long)NN * 8);
        L1D(k_fill_f32, (long)NN * N8, out_v, 0.f, (long)NN * N8);
        L1D(k_fill_f32, (long)NN * 128, S, 0.f, (long)NN * 128);
        float scale = 1.0f / sqrtf((float)dout);
        L1D(k_alpha, (long)NE * HEADS, qb, kb, Qe, eemb, src, dstp, alpha, mkey, dout, scale);
        L1D(k_expsum, (long)NE * HEADS, alpha, mkey, denom, dstp);
        L1D(k_scatter, (long)NE * HEADS, alpha, denom, vb, eemb, src, dstp, out_v, S, dout);

        // out_v += S @ We (per-head WMMA GEMM, K=32, accumulate in place)
        L1D(k_pack_heads_f16, (long)NN * HEADS * 32, S, Sf16, 16, 32, (long)NN * HEADS * 32);
        for (int h = 0; h < HEADS; ++h)
            k_wmma_gemm<<<dim3(NN / 16, dout / 16), 32, 0, stream>>>(
                Sf16 + h * 32, HEADS * 32, WeT + h * dout * 32, 32,
                nullptr, out_v + h * dout, N8, out_v + h * dout, N8, 32);

        // h = relu(mean_heads + skip) + res
        L1D(k_finalize, (long)NN * dout, out_v, skip, res, hB, dout);
        { float* t = hA; hA = hB; hB = t; }
    }

    // global mean pool + MLP head
    L1D(k_fill_f32, (long)NG * 112, g, 0.f, (long)NG * 112);
    L1D(k_fill_f32, (long)NG, cnt, 0.f, (long)NG);
    L1D(k_pool, (long)NN * 112, hA, batch, g, cnt, 112);
    L1D(k_head1, NG * NFF, g, cnt, headW1, headb1, ffb, 112);
    k_head2<<<1, 32, 0, stream>>>(ffb, headW2, headb2, out);
}
